// DeltaBlock_30485677867998
// MI455X (gfx1250) — compile-verified, matrix operand order checked by reading, not run on hardware
//
#include <hip/hip_runtime.h>
#include <hip/hip_bf16.h>

// ---------------------------------------------------------------------------
// DeltaNet chunked delta rule for MI455X (gfx1250).
//   B=8, L=4096, D=1024, C=64.  All matmuls on v_wmma_f32_16x16x32_bf16.
//   Scan keeps a 1024x32 f32 state slab + bf16 transposed mirror resident in
//   CDNA5's 320KB LDS (196KB used); 8 batches x 32 slabs = 256 blocks.
//   chunk_prep stores W negated so u = U0 - W@S is a plain WMMA accumulation
//   seeded with U0 (bf16 WMMA has no neg_a modifier).
// ---------------------------------------------------------------------------

#define B_  8
#define L_  4096
#define D_  1024
#define C_  64
#define NC_ (L_ / C_)
#define M_  (B_ * L_)
#define NSL 32          // state-slab width (columns of S per block)

typedef __bf16 bf16_t;
typedef __attribute__((ext_vector_type(16))) __bf16 v16bf;
typedef __attribute__((ext_vector_type(8)))  float  v8f;

static __device__ __forceinline__ int lane_id() { return threadIdx.x & 31; }
static __device__ __forceinline__ int wave_id() { return threadIdx.x >> 5; }

// D = A*B + C  (16x16x32 bf16 -> f32)
static __device__ __forceinline__ v8f wmma_acc(v16bf a, v16bf b, v8f c) {
  return __builtin_amdgcn_wmma_f32_16x16x32_bf16(false, a, false, b, (short)0, c,
                                                 false, false);
}

// Fragment from a row-major 16xK tile (global or LDS): each lane grabs 16
// contiguous bf16 (32B vector load).  Works for A-tiles and for B-tiles whose
// source is stored N-major (row = N, contiguous K).
static __device__ __forceinline__ v16bf frag_rm(const bf16_t* p, int ld,
                                                int k0) {
  const int r = lane_id() & 15;
  const int h = lane_id() >> 4;
  return *(const v16bf*)(p + (size_t)r * ld + k0 + h * 16);
}

// Fragment where the source is K-major with row stride ldk (column gather).
static __device__ __forceinline__ v16bf fragB_strided_bf16(
    const bf16_t* __restrict__ p, int ldk, int k0) {
  const int n  = lane_id() & 15;
  const int kb = k0 + ((lane_id() >> 4) << 4);
  v16bf v;
#pragma unroll
  for (int i = 0; i < 16; ++i) v[i] = p[(size_t)(kb + i) * ldk + n];
  return v;
}

// C/D fragment row for element i:  M = i + 8*(lane/16), N = lane%16.
static __device__ __forceinline__ int acc_row(int i) {
  return i + ((lane_id() >> 4) << 3);
}
static __device__ __forceinline__ int acc_col() { return lane_id() & 15; }

// ---------------------------------------------------------------------------
// f32 -> bf16 elementwise convert
// ---------------------------------------------------------------------------
__global__ void cvt_f32_bf16(const float* __restrict__ in,
                             bf16_t* __restrict__ out, int n) {
  int i = blockIdx.x * blockDim.x + threadIdx.x;
  if (i < n) out[i] = (bf16_t)in[i];
}

// ---------------------------------------------------------------------------
// GEMM:  out[m][n] = sum_k A[m][k] * Wt[n][k] + bias[n]
//   Block tile 128x128 (8 waves), wave tile 64x32: 8 WMMA per 6 fragment
//   loads.  Double-buffered, #pragma unroll 1 keeps registers bounded.
// ---------------------------------------------------------------------------
template <bool STORE_F32>
__global__ void gemm_bias(const bf16_t* __restrict__ A,
                          const bf16_t* __restrict__ Wt,
                          const float* __restrict__ bias, void* __restrict__ out,
                          int M, int N, int K) {
  const int mh = wave_id() >> 2;  // 0..1
  const int nw = wave_id() & 3;   // 0..3
  const int m0 = blockIdx.x * 128 + mh * 64;
  const int n0 = blockIdx.y * 128 + nw * 32;

  v8f acc[4][2];
#pragma unroll
  for (int t = 0; t < 4; ++t)
#pragma unroll
    for (int j = 0; j < 2; ++j) acc[t][j] = (v8f){};

  const bf16_t* pA = A + (size_t)m0 * K;
  const bf16_t* pB = Wt + (size_t)n0 * K;

  v16bf ac[4], bc[2];
#pragma unroll
  for (int t = 0; t < 4; ++t) ac[t] = frag_rm(pA + (size_t)t * 16 * K, K, 0);
#pragma unroll
  for (int j = 0; j < 2; ++j) bc[j] = frag_rm(pB + (size_t)j * 16 * K, K, 0);

#pragma unroll 1
  for (int k0 = 0; k0 < K; k0 += 32) {
    const int kn = (k0 + 32 < K) ? (k0 + 32) : k0;
    __builtin_prefetch(pB + kn + 64, 0, 0);  // global_prefetch_b8 path
    v16bf an[4], bn[2];
#pragma unroll
    for (int t = 0; t < 4; ++t) an[t] = frag_rm(pA + (size_t)t * 16 * K, K, kn);
#pragma unroll
    for (int j = 0; j < 2; ++j) bn[j] = frag_rm(pB + (size_t)j * 16 * K, K, kn);
#pragma unroll
    for (int t = 0; t < 4; ++t)
#pragma unroll
      for (int j = 0; j < 2; ++j)
        acc[t][j] = wmma_acc(ac[t], bc[j], acc[t][j]);
#pragma unroll
    for (int t = 0; t < 4; ++t) ac[t] = an[t];
#pragma unroll
    for (int j = 0; j < 2; ++j) bc[j] = bn[j];
  }

#pragma unroll
  for (int j = 0; j < 2; ++j) {
    const int col = n0 + j * 16 + acc_col();
    const float bv = bias[col];
#pragma unroll
    for (int t = 0; t < 4; ++t) {
#pragma unroll
      for (int i = 0; i < 8; ++i) {
        const int row = m0 + t * 16 + acc_row(i);
        const float v = acc[t][j][i] + bv;
        if (STORE_F32)
          ((float*)out)[(size_t)row * N + col] = v;
        else
          ((bf16_t*)out)[(size_t)row * N + col] = (bf16_t)v;
      }
    }
  }
}

// ---------------------------------------------------------------------------
// beta[m] = 1 / (sum_k K[m,k]^2 + 1e-6)   -- one wave per row
// ---------------------------------------------------------------------------
__global__ void beta_kernel(const bf16_t* __restrict__ Km,
                            float* __restrict__ beta) {
  const int m = blockIdx.x * 8 + wave_id();
  float s = 0.f;
  for (int k = lane_id(); k < D_; k += 32) {
    float v = (float)Km[(size_t)m * D_ + k];
    s += v * v;
  }
#pragma unroll
  for (int off = 16; off > 0; off >>= 1) s += __shfl_down(s, off, 32);
  if (lane_id() == 0) beta[m] = 1.f / (s + 1e-6f);
}

// ---------------------------------------------------------------------------
// Per-(batch,chunk) prep:  T = fwdsub-inv of (I + tril(beta*K K^T)),
//   Wm = -(T*beta) @ K   (negated!),  U0 = (T*beta) @ V,
//   A  = (Q K^T) . tril_incl
// ---------------------------------------------------------------------------
__global__ void chunk_prep(const bf16_t* __restrict__ Q,
                           const bf16_t* __restrict__ Km,
                           const bf16_t* __restrict__ V,
                           const float* __restrict__ beta,
                           bf16_t* __restrict__ Wm, bf16_t* __restrict__ U0,
                           bf16_t* __restrict__ Amat) {
  __shared__ float  Tl[64][65];
  __shared__ bf16_t Tb[64][64];

  const int bc = blockIdx.x;
  const int b = bc / NC_, c = bc % NC_;
  const size_t row0 = (size_t)b * L_ + (size_t)c * C_;
  const bf16_t* Qc = Q + row0 * D_;
  const bf16_t* Kc = Km + row0 * D_;
  const bf16_t* Vc = V + row0 * D_;
  const float*  bt = beta + row0;

  // ---- stage 1: KK = K @ K^T ;  Tl = -beta_i * KK strictly-lower ----------
  for (int tile = wave_id(); tile < 16; tile += 8) {
    const int mt = tile >> 2, nt = tile & 3;
    v8f acc = (v8f){};
#pragma unroll 1
    for (int k0 = 0; k0 < D_; k0 += 32) {
      v16bf a = frag_rm(Kc + (size_t)(mt * 16) * D_, D_, k0);
      v16bf bb = frag_rm(Kc + (size_t)(nt * 16) * D_, D_, k0);  // rows of K
      acc = wmma_acc(a, bb, acc);
    }
#pragma unroll
    for (int i = 0; i < 8; ++i) {
      const int r = mt * 16 + acc_row(i);
      const int cc = nt * 16 + acc_col();
      const float nb = -bt[r];                // unconditional load -> cndmask
      const float prod = nb * acc[i];
      Tl[r][cc] = (r > cc) ? prod : 0.f;
    }
  }
  __syncthreads();

  // ---- stage 2: forward substitution (serial over 64 rows, tiny) ----------
  for (int i = 1; i < 64; ++i) {
    float t = 0.f;
    const int j = threadIdx.x;
    if (j < i) {
      for (int k = 0; k < i; ++k) t += Tl[i][k] * Tl[k][j];
    }
    __syncthreads();
    if (j < i) Tl[i][j] += t;
    __syncthreads();
  }

  // ---- stage 3: T += I, fold beta into columns, cast to bf16 --------------
  for (int idx = threadIdx.x; idx < 64 * 64; idx += blockDim.x) {
    const int i = idx >> 6, t = idx & 63;
    float v = Tl[i][t] + ((i == t) ? 1.f : 0.f);
    Tb[i][t] = (bf16_t)(v * bt[t]);
  }
  __syncthreads();

  // ---- stage 4: Wm = -(Tb @ K), U0 = Tb @ V  (64x1024 each) ---------------
#pragma unroll 1
  for (int tile = wave_id(); tile < 512; tile += 8) {
    const bool isU = tile >= 256;
    const int tt = tile & 255;
    const int mt = tt >> 6, nt = tt & 63;
    const bf16_t* src = isU ? Vc : Kc;
    bf16_t* dst = (isU ? U0 : Wm) + row0 * D_;
    const float sgn = isU ? 1.f : -1.f;  // store -W so scan can accumulate
    v8f acc = (v8f){};
#pragma unroll
    for (int kt = 0; kt < 2; ++kt) {
      v16bf a = frag_rm(&Tb[mt * 16][0], 64, kt * 32);
      v16bf bb = fragB_strided_bf16(src + nt * 16, D_, kt * 32);
      acc = wmma_acc(a, bb, acc);
    }
#pragma unroll
    for (int i = 0; i < 8; ++i) {
      const int r = mt * 16 + acc_row(i);
      const int cc = nt * 16 + acc_col();
      dst[(size_t)r * D_ + cc] = (bf16_t)(sgn * acc[i]);
    }
  }

  // ---- stage 5: A = (Q @ K^T) masked to inclusive-lower -------------------
  bf16_t* Ac = Amat + (size_t)bc * (C_ * C_);
  for (int tile = wave_id(); tile < 16; tile += 8) {
    const int mt = tile >> 2, nt = tile & 3;
    v8f acc = (v8f){};
#pragma unroll 1
    for (int k0 = 0; k0 < D_; k0 += 32) {
      v16bf a = frag_rm(Qc + (size_t)(mt * 16) * D_, D_, k0);
      v16bf bb = frag_rm(Kc + (size_t)(nt * 16) * D_, D_, k0);
      acc = wmma_acc(a, bb, acc);
    }
#pragma unroll
    for (int i = 0; i < 8; ++i) {
      const int r = mt * 16 + acc_row(i);
      const int cc = nt * 16 + acc_col();
      Ac[r * 64 + cc] = (bf16_t)((r >= cc) ? acc[i] : 0.f);
    }
  }
}

// ---------------------------------------------------------------------------
// Scan: block = (slab, batch).  LDS layout:
//   Sl[1024][32] f32   -- master state slab (accumulated in f32)
//   St[32][1024] bf16  -- transposed bf16 mirror: B-fragments = vector loads
//   ut[32][64]   bf16  -- transposed u: B-fragments = vector loads
// Per chunk:  u = U0 + (-W)@S ;  o = Q@S + A@u ;  S += K^T@u.
// ---------------------------------------------------------------------------
__global__ void scan_kernel(const bf16_t* __restrict__ Q,
                            const bf16_t* __restrict__ Km,
                            const bf16_t* __restrict__ Wm,
                            const bf16_t* __restrict__ U0,
                            const bf16_t* __restrict__ Amat,
                            bf16_t* __restrict__ Ob) {
  extern __shared__ char smem[];
  float*  Sl = (float*)smem;                                  // [1024][NSL]
  bf16_t* St = (bf16_t*)(smem + (size_t)D_ * NSL * 4);        // [NSL][1024]
  bf16_t* ut = (bf16_t*)(smem + (size_t)D_ * NSL * 4 +
                         (size_t)NSL * D_ * 2);               // [NSL][64]

  const int slab = blockIdx.x;     // 0..31
  const int b    = blockIdx.y;     // 0..7
  const int nsl  = slab * NSL;

  for (int i = threadIdx.x; i < D_ * NSL; i += blockDim.x) Sl[i] = 0.f;
  for (int i = threadIdx.x; i < NSL * D_; i += blockDim.x) St[i] = (bf16_t)0.f;
  __syncthreads();

#pragma unroll 1
  for (int c = 0; c < NC_; ++c) {
    const size_t row0 = (size_t)b * L_ + (size_t)c * C_;
    const bf16_t* Qc = Q + row0 * D_;
    const bf16_t* Kc = Km + row0 * D_;
    const bf16_t* Wc = Wm + row0 * D_;
    const bf16_t* Uc = U0 + row0 * D_;
    const bf16_t* Ac = Amat + ((size_t)b * NC_ + c) * (C_ * C_);

    // ---- u = U0 + (-W) @ S : 8 tiles, one per wave ------------------------
    {
      const int mt = wave_id() >> 1, nt = wave_id() & 1;
      v8f acc;
#pragma unroll
      for (int i = 0; i < 8; ++i) {
        const int t = mt * 16 + acc_row(i);
        acc[i] = (float)Uc[(size_t)t * D_ + nsl + nt * 16 + acc_col()];
      }
#pragma unroll 2
      for (int k0 = 0; k0 < D_; k0 += 32) {
        v16bf a = frag_rm(Wc + (size_t)(mt * 16) * D_, D_, k0);
        v16bf bb = frag_rm(St + (size_t)(nt * 16) * D_, D_, k0);  // LDS vec
        acc = wmma_acc(a, bb, acc);
      }
#pragma unroll
      for (int i = 0; i < 8; ++i) {
        const int t = mt * 16 + acc_row(i);
        ut[(nt * 16 + acc_col()) * 64 + t] = (bf16_t)acc[i];
      }
    }
    __syncthreads();

    // ---- o = Q @ S + A @ u : 8 tiles, one per wave ------------------------
    {
      const int mt = wave_id() >> 1, nt = wave_id() & 1;
      v8f acc = (v8f){};
#pragma unroll 2
      for (int k0 = 0; k0 < D_; k0 += 32) {
        v16bf a = frag_rm(Qc + (size_t)(mt * 16) * D_, D_, k0);
        v16bf bb = frag_rm(St + (size_t)(nt * 16) * D_, D_, k0);
        acc = wmma_acc(a, bb, acc);
      }
#pragma unroll
      for (int kt = 0; kt < 2; ++kt) {
        v16bf a = frag_rm(Ac + mt * 16 * 64, 64, kt * 32);
        v16bf bb = frag_rm(ut + nt * 16 * 64, 64, kt * 32);
        acc = wmma_acc(a, bb, acc);
      }
#pragma unroll
      for (int i = 0; i < 8; ++i) {
        const int t = mt * 16 + acc_row(i);
        Ob[(row0 + t) * D_ + nsl + nt * 16 + acc_col()] = (bf16_t)acc[i];
      }
    }
    __syncthreads();

    // ---- S += K^T @ u : 128 tiles, 16 per wave ----------------------------
#pragma unroll 1
    for (int tile = wave_id(); tile < 128; tile += 8) {
      const int mt = tile >> 1, nt = tile & 1;  // mt: S row-tile (0..63)
      v8f acc;
#pragma unroll
      for (int i = 0; i < 8; ++i)
        acc[i] = Sl[(mt * 16 + acc_row(i)) * NSL + nt * 16 + acc_col()];
#pragma unroll
      for (int kt = 0; kt < 2; ++kt) {
        v16bf a = fragB_strided_bf16(Kc + mt * 16, D_, kt * 32);  // K^T rows
        v16bf bb = frag_rm(ut + nt * 16 * 64, 64, kt * 32);
        acc = wmma_acc(a, bb, acc);
      }
#pragma unroll
      for (int i = 0; i < 8; ++i) {
        const int r = mt * 16 + acc_row(i);
        const int cc = nt * 16 + acc_col();
        Sl[r * NSL + cc] = acc[i];
        St[(size_t)cc * D_ + r] = (bf16_t)acc[i];
      }
    }
    __syncthreads();
  }
}

// ---------------------------------------------------------------------------
extern "C" void kernel_launch(void* const* d_in, const int* in_sizes, int n_in,
                              void* d_out, int out_size, void* d_ws,
                              size_t ws_size, hipStream_t stream) {
  const float* X    = (const float*)d_in[0];
  const float* Wq_w = (const float*)d_in[2];
  const float* Wq_b = (const float*)d_in[3];
  const float* Wk_w = (const float*)d_in[4];
  const float* Wk_b = (const float*)d_in[5];
  const float* Wv_w = (const float*)d_in[6];
  const float* Wv_b = (const float*)d_in[7];
  const float* Wo_w = (const float*)d_in[8];
  const float* Wo_b = (const float*)d_in[9];

  size_t off = 0;
  auto carve = [&](size_t bytes) -> void* {
    void* p = (char*)d_ws + off;
    off += (bytes + 255) & ~(size_t)255;
    return p;
  };
  bf16_t* Xb   = (bf16_t*)carve((size_t)M_ * D_ * 2);
  bf16_t* Wqb  = (bf16_t*)carve((size_t)D_ * D_ * 2);
  bf16_t* Wkb  = (bf16_t*)carve((size_t)D_ * D_ * 2);
  bf16_t* Wvb  = (bf16_t*)carve((size_t)D_ * D_ * 2);
  bf16_t* Wob  = (bf16_t*)carve((size_t)D_ * D_ * 2);
  bf16_t* Qb   = (bf16_t*)carve((size_t)M_ * D_ * 2);
  bf16_t* Kb   = (bf16_t*)carve((size_t)M_ * D_ * 2);
  bf16_t* Vb   = (bf16_t*)carve((size_t)M_ * D_ * 2);
  float*  beta = (float*)carve((size_t)M_ * 4);
  bf16_t* Wm   = (bf16_t*)carve((size_t)M_ * D_ * 2);
  bf16_t* U0   = (bf16_t*)carve((size_t)M_ * D_ * 2);
  bf16_t* Amat = (bf16_t*)carve((size_t)B_ * NC_ * C_ * C_ * 2);
  bf16_t* Ob   = (bf16_t*)carve((size_t)M_ * D_ * 2);
  (void)ws_size; (void)n_in; (void)in_sizes; (void)out_size;

  const int nMD = M_ * D_, nDD = D_ * D_;
  cvt_f32_bf16<<<(nMD + 255) / 256, 256, 0, stream>>>(X, Xb, nMD);
  cvt_f32_bf16<<<(nDD + 255) / 256, 256, 0, stream>>>(Wq_w, Wqb, nDD);
  cvt_f32_bf16<<<(nDD + 255) / 256, 256, 0, stream>>>(Wk_w, Wkb, nDD);
  cvt_f32_bf16<<<(nDD + 255) / 256, 256, 0, stream>>>(Wv_w, Wvb, nDD);
  cvt_f32_bf16<<<(nDD + 255) / 256, 256, 0, stream>>>(Wo_w, Wob, nDD);

  dim3 gg(M_ / 128, D_ / 128);
  gemm_bias<false><<<gg, 256, 0, stream>>>(Xb, Wqb, Wq_b, Qb, M_, D_, D_);
  gemm_bias<false><<<gg, 256, 0, stream>>>(Xb, Wkb, Wk_b, Kb, M_, D_, D_);
  gemm_bias<false><<<gg, 256, 0, stream>>>(Xb, Wvb, Wv_b, Vb, M_, D_, D_);

  beta_kernel<<<M_ / 8, 256, 0, stream>>>(Kb, beta);

  chunk_prep<<<B_ * NC_, 256, 0, stream>>>(Qb, Kb, Vb, beta, Wm, U0, Amat);

  const size_t smem = (size_t)D_ * NSL * 4 + (size_t)NSL * D_ * 2 +
                      (size_t)NSL * 64 * 2;  // 192 KB + 4 KB
  scan_kernel<<<dim3(D_ / NSL, B_), 256, smem, stream>>>(Qb, Kb, Wm, U0, Amat,
                                                         Ob);

  gemm_bias<true><<<gg, 256, 0, stream>>>(Ob, Wob, Wo_b, (float*)d_out, M_, D_,
                                          D_);
}
